// TenNet_Tag_20538533610035
// MI455X (gfx1250) — matrix operand order repaired and checked by reading, not compiled, not run on hardware
//
#include <hip/hip_runtime.h>
#include <hip/hip_bf16.h>

// ---------------------------------------------------------------------------
// TenNet on gfx1250: sparsity-aware front end + WMMA (f16 in, f32 acc) convs.
// Round 3: fix async-to-LDS builtin signatures (int / int4 pointers, AS1/AS3).
// ---------------------------------------------------------------------------

typedef __attribute__((ext_vector_type(16))) _Float16 v16h;
typedef __attribute__((ext_vector_type(8)))  float    v8f;
typedef __attribute__((ext_vector_type(4)))  int      vi4;

#define NB  16      // batch
#define NV  10000   // vocab / image H at input
#define ND  300     // emb dim / image W at input
#define NT  50      // tags per batch
#define KPAD_MAX 128

#if defined(__HIP_DEVICE_COMPILE__) && __has_builtin(__builtin_amdgcn_global_load_async_to_lds_b128) && __has_builtin(__builtin_amdgcn_global_load_async_to_lds_b32)
#define HAVE_ASYNC_LDS 1
#else
#define HAVE_ASYNC_LDS 0
#endif

// -------------------------- flags (sparsity bitmap) ------------------------

__global__ void zero_flags_kernel(unsigned char* f, int n) {
    int i = blockIdx.x * blockDim.x + threadIdx.x;
    if (i < n) f[i] = 0;
}

__global__ void scatter_flags_kernel(const int* __restrict__ tags, unsigned char* __restrict__ f) {
    int i = blockIdx.x * blockDim.x + threadIdx.x;
    if (i < NB * NT) {
        int b = i / NT;
        int t = tags[i];
        f[b * NV + t] = 1;
    }
}

// ------------- fused scatter + conv1(2->4,3x3,pad1) + relu + pool(3,3,1) ---

__global__ void scatter_conv1_pool_kernel(const unsigned char* __restrict__ flags,
                                          const float* __restrict__ emb,
                                          const float* __restrict__ w,   // [4][2][3][3]
                                          const float* __restrict__ bias,// [4]
                                          float* __restrict__ out,       // [B,4,H1,W1]
                                          int H1, int W1) {
    int idx = blockIdx.x * blockDim.x + threadIdx.x;
    int total = NB * 4 * H1 * W1;
    if (idx >= total) return;
    int pw = idx % W1;
    int t  = idx / W1;
    int ph = t % H1;  t /= H1;
    int oc = t % 4;
    int b  = t / 4;

    float we[3][3];
    #pragma unroll
    for (int dy = 0; dy < 3; ++dy)
        #pragma unroll
        for (int dx = 0; dx < 3; ++dx)
            we[dy][dx] = w[((oc * 2 + 0) * 3 + dy) * 3 + dx] +
                         w[((oc * 2 + 1) * 3 + dy) * 3 + dx];
    float bb = bias[oc];
    const unsigned char* fl = flags + (size_t)b * NV;

    float mx = -3.0e38f;
    #pragma unroll
    for (int py = 0; py < 3; ++py) {
        int y = 3 * ph - 1 + py;
        if (y < 0 || y >= NV) continue;
        #pragma unroll
        for (int px = 0; px < 3; ++px) {
            int x = 3 * pw - 1 + px;
            if (x < 0 || x >= ND) continue;
            float s = bb;
            #pragma unroll
            for (int dy = 0; dy < 3; ++dy) {
                int r = y + dy - 1;
                if (r < 0 || r >= NV) continue;
                if (!fl[r]) continue;                 // background row: skip
                const float* er = emb + (size_t)r * ND;
                #pragma unroll
                for (int dx = 0; dx < 3; ++dx) {
                    int c = x + dx - 1;
                    if (c < 0 || c >= ND) continue;
                    s += we[dy][dx] * er[c];
                }
            }
            s  = fmaxf(s, 0.0f);                      // relu
            mx = fmaxf(mx, s);                        // maxpool
        }
    }
    out[idx] = mx;
}

// ------------------- generic implicit-GEMM conv via WMMA -------------------
// One wave per block; each wave computes a 32(M) x 16(N) output tile as two
// 16x16 WMMAs sharing one B fragment. K = IC*KH*KW padded to 32.
// LDS holds: f16 weights [16 x Kpad], and a k-decode table
//   ktab[k] = { ic*HW + dyp*W + dxp,  (dyp<<16)|(dxp&0xffff) }
// so the A-gather address is simply  in[p + ktab[k].x]  (p = y*W + x),
// with bounds checked from the packed deltas.

__global__ __launch_bounds__(32)
void conv_wmma_kernel(const float* __restrict__ in,   // [B,IC,H,W]
                      const float* __restrict__ wgt,  // [OC,IC,KH,KW]
                      const float* __restrict__ bias, // [OC]
                      float* __restrict__ out,        // [B,OCtot,H,W] (+oc0)
                      int IC, int H, int W, int OC, int KH, int KW, int pad,
                      int OCtot, int oc0, int relu) {
    const int lane = threadIdx.x;
    const int b    = blockIdx.y;
    const int HW   = H * W;
    const int K    = IC * KH * KW;
    const int Kpad = (K + 31) & ~31;

    __shared__ _Float16 wlds[16 * KPAD_MAX];   // 4 KB
    __shared__ int2     ktab[KPAD_MAX];        // 1 KB
#if HAVE_ASYNC_LDS
    __shared__ __attribute__((aligned(16))) float wldsf[16 * KPAD_MAX];  // 8 KB raw f32
#endif

    // ---- k -> (ic,dy,dx) decode table (one idiv chain per k, not per use) ----
    for (int k = lane; k < Kpad; k += 32) {
        if (k < K) {
            int ic = k / (KH * KW);
            int r  = k % (KH * KW);
            int dyp = r / KW - pad;
            int dxp = r % KW - pad;
            ktab[k] = make_int2(ic * HW + dyp * W + dxp,
                                (dyp << 16) | (dxp & 0xFFFF));
        } else {
            ktab[k] = make_int2(0, 0x40004000);   // sentinel: always OOB -> 0
        }
    }

    // ---- weight staging into LDS ----
#if HAVE_ASYNC_LDS
    {
        const int nw  = OC * K;
        const int nw4 = nw & ~3;
        for (int i = lane * 4; i < nw4; i += 32 * 4) {
            __builtin_amdgcn_global_load_async_to_lds_b128(
                (__attribute__((address_space(1))) vi4*)(wgt + i),
                (__attribute__((address_space(3))) vi4*)(wldsf + i), 0, 0);
        }
        for (int i = nw4 + lane; i < nw; i += 32) {
            __builtin_amdgcn_global_load_async_to_lds_b32(
                (__attribute__((address_space(1))) int*)(wgt + i),
                (__attribute__((address_space(3))) int*)(wldsf + i), 0, 0);
        }
#if __has_builtin(__builtin_amdgcn_s_wait_asynccnt)
        __builtin_amdgcn_s_wait_asynccnt(0);
#else
        asm volatile("s_wait_asynccnt 0" ::: "memory");
#endif
        __syncthreads();
        for (int i = lane; i < 16 * Kpad; i += 32) {
            int n = i / Kpad, k = i % Kpad;
            wlds[i] = (n < OC && k < K) ? (_Float16)wldsf[n * K + k] : (_Float16)0.0f;
        }
    }
#else
    for (int i = lane; i < 16 * Kpad; i += 32) {
        int n = i / Kpad, k = i % Kpad;
        wlds[i] = (n < OC && k < K) ? (_Float16)wgt[n * K + k] : (_Float16)0.0f;
    }
#endif
    __syncthreads();

    const int mrow = lane & 15;
    const int ncol = lane & 15;
    const int hiA  = (lane >= 16) ? 8 : 0;
    const int hiB  = (lane >= 16) ? 16 : 0;
    const int p0   = blockIdx.x * 32 + mrow;    // M sub-tile 0
    const int p1   = p0 + 16;                   // M sub-tile 1
    const int y0   = (p0 < HW) ? p0 / W : 0;
    const int x0   = (p0 < HW) ? p0 % W : 0;
    const int y1   = (p1 < HW) ? p1 / W : 0;
    const int x1   = (p1 < HW) ? p1 % W : 0;
    const float* inb = in + (size_t)b * IC * HW;
    __builtin_prefetch(inb + p0, 0, 0);         // global_prefetch_b8

    v8f acc0 = {}, acc1 = {};
    for (int k0 = 0; k0 < Kpad; k0 += 32) {
        v16h bf;
        #pragma unroll
        for (int e = 0; e < 16; ++e)
            bf[e] = wlds[ncol * Kpad + k0 + ((e < 8) ? 0 : 16) + hiA + (e & 7)];

        v16h a0, a1;
        #pragma unroll
        for (int e = 0; e < 16; ++e) {
            int kl  = ((e < 8) ? 0 : 16) + hiA + (e & 7);
            int2 kt = ktab[k0 + kl];
            int dyp = kt.y >> 16;
            int dxp = (int)(short)(kt.y & 0xFFFF);

            float v0 = 0.0f, v1 = 0.0f;
            {
                int iy = y0 + dyp, ix = x0 + dxp;
                if (p0 < HW && iy >= 0 && iy < H && ix >= 0 && ix < W)
                    v0 = inb[p0 + kt.x];
            }
            {
                int iy = y1 + dyp, ix = x1 + dxp;
                if (p1 < HW && iy >= 0 && iy < H && ix >= 0 && ix < W)
                    v1 = inb[p1 + kt.x];
            }
            a0[e] = (_Float16)v0;
            a1[e] = (_Float16)v1;
        }
        acc0 = __builtin_amdgcn_wmma_f32_16x16x32_f16(
                   false, a0, false, bf, (short)0, acc0, false, false);
        acc1 = __builtin_amdgcn_wmma_f32_16x16x32_f16(
                   false, a1, false, bf, (short)0, acc1, false, false);
    }

    if (ncol < OC) {
        float bv = bias[ncol];
        float* ob = out + ((size_t)b * OCtot + oc0 + ncol) * HW;
        #pragma unroll
        for (int r = 0; r < 8; ++r) {
            int pm = blockIdx.x * 32 + r + hiA;
            if (pm < HW) {
                float v = acc0[r] + bv;
                if (relu) v = fmaxf(v, 0.0f);
                ob[pm] = v;
            }
            int pm1 = pm + 16;
            if (pm1 < HW) {
                float v = acc1[r] + bv;
                if (relu) v = fmaxf(v, 0.0f);
                ob[pm1] = v;
            }
        }
    }
}

// ----------------------------- 3x3 maxpool, pad=1 --------------------------

__global__ void maxpool3_kernel(const float* __restrict__ in, float* __restrict__ out,
                                int BC, int H, int W, int Ho, int Wo, int stride) {
    int idx = blockIdx.x * blockDim.x + threadIdx.x;
    int total = BC * Ho * Wo;
    if (idx >= total) return;
    int wo = idx % Wo;
    int t  = idx / Wo;
    int ho = t % Ho;
    int bc = t / Ho;
    const float* ip = in + (size_t)bc * H * W;
    int y0 = ho * stride - 1, x0 = wo * stride - 1;
    float mx = -3.0e38f;
    #pragma unroll
    for (int dy = 0; dy < 3; ++dy) {
        int y = y0 + dy;
        if (y < 0 || y >= H) continue;
        #pragma unroll
        for (int dx = 0; dx < 3; ++dx) {
            int x = x0 + dx;
            if (x < 0 || x >= W) continue;
            mx = fmaxf(mx, ip[(size_t)y * W + x]);
        }
    }
    out[idx] = mx;
}

// ------------------------------ global avg pool ----------------------------

__global__ void gap_kernel(const float* __restrict__ in, float* __restrict__ out, int HW) {
    __shared__ float red[256];
    int bc = blockIdx.x;
    const float* ip = in + (size_t)bc * HW;
    float s = 0.0f;
    for (int i = threadIdx.x; i < HW; i += 256) s += ip[i];
    red[threadIdx.x] = s;
    __syncthreads();
    for (int off = 128; off > 0; off >>= 1) {
        if (threadIdx.x < off) red[threadIdx.x] += red[threadIdx.x + off];
        __syncthreads();
    }
    if (threadIdx.x == 0) out[bc] = red[0] / (float)HW;
}

// -------------------------- FC: [16,44] x [128,44]^T -----------------------

__global__ __launch_bounds__(32)
void fc_wmma_kernel(const float* __restrict__ inp,  // [16,44]
                    const float* __restrict__ w,    // [128,44]
                    const float* __restrict__ bias, // [128]
                    float* __restrict__ out) {      // [16,128]
    const int lane = threadIdx.x;
    const int nt   = blockIdx.x;                    // 0..7 -> 16 outputs each
    const int K = 44, Kpad = 64;
    const int mrow = lane & 15;
    const int ncol = lane & 15;
    const int hiA  = (lane >= 16) ? 8 : 0;
    const int hiB  = (lane >= 16) ? 16 : 0;
    const int gn   = nt * 16 + ncol;

    v8f acc = {};
    for (int k0 = 0; k0 < Kpad; k0 += 32) {
        v16h a, bf;
        #pragma unroll
        for (int e = 0; e < 16; ++e) {
            int ka = k0 + ((e < 8) ? 0 : 16) + hiA + (e & 7);
            a[e] = (ka < K) ? (_Float16)inp[mrow * K + ka] : (_Float16)0.0f;
            int kb = k0 + ((e < 8) ? 0 : 16) + hiA + (e & 7);
            bf[e] = (kb < K) ? (_Float16)w[gn * K + kb] : (_Float16)0.0f;
        }
        acc = __builtin_amdgcn_wmma_f32_16x16x32_f16(
                  false, a, false, bf, (short)0, acc, false, false);
    }
    float bv = bias[gn];
    #pragma unroll
    for (int r = 0; r < 8; ++r) {
        int row = r + hiA;
        out[row * 128 + gn] = acc[r] + bv;
    }
}

// ---------------------------------------------------------------------------
// host driver
// ---------------------------------------------------------------------------

extern "C" void kernel_launch(void* const* d_in, const int* in_sizes, int n_in,
                              void* d_out, int out_size, void* d_ws, size_t ws_size,
                              hipStream_t stream) {
    (void)in_sizes; (void)n_in; (void)out_size; (void)ws_size;

    const int* tags     = (const int*)d_in[0];
    const float* emb    = (const float*)d_in[1];
    const float* c1_w   = (const float*)d_in[2],  *c1_b   = (const float*)d_in[3];
    const float* c2_w   = (const float*)d_in[4],  *c2_b   = (const float*)d_in[5];
    const float* c3_w   = (const float*)d_in[6],  *c3_b   = (const float*)d_in[7];
    const float* i1p1w  = (const float*)d_in[8],  *i1p1b  = (const float*)d_in[9];
    const float* i1p2aw = (const float*)d_in[10], *i1p2ab = (const float*)d_in[11];
    const float* i1p2bw = (const float*)d_in[12], *i1p2bb = (const float*)d_in[13];
    const float* i1p3aw = (const float*)d_in[14], *i1p3ab = (const float*)d_in[15];
    const float* i1p3bw = (const float*)d_in[16], *i1p3bb = (const float*)d_in[17];
    const float* i1p4w  = (const float*)d_in[18], *i1p4b  = (const float*)d_in[19];
    const float* i2p1w  = (const float*)d_in[20], *i2p1b  = (const float*)d_in[21];
    const float* i2p2aw = (const float*)d_in[22], *i2p2ab = (const float*)d_in[23];
    const float* i2p2bw = (const float*)d_in[24], *i2p2bb = (const float*)d_in[25];
    const float* i2p3aw = (const float*)d_in[26], *i2p3ab = (const float*)d_in[27];
    const float* i2p3bw = (const float*)d_in[28], *i2p3bb = (const float*)d_in[29];
    const float* i2p4w  = (const float*)d_in[30], *i2p4b  = (const float*)d_in[31];
    const float* fc_w   = (const float*)d_in[32], *fc_b   = (const float*)d_in[33];

    const int B = NB;
    const int H1 = 3334, W1 = 100;   // after pool1 (10000x300, k3 s3 p1)
    const int H2 = 1112, W2 = 34;    // after pool2
    const int H3 = 556,  W3 = 17;    // after pool3

    // ---- workspace layout (bump allocation with reuse) ----
    unsigned char* flags = (unsigned char*)d_ws;                 // B*V = 160000 B
    float* fb = (float*)((char*)d_ws + 160000);                  // float arena

    const size_t nPool1 = (size_t)B * 4  * H1 * W1;
    const size_t nC2    = (size_t)B * 8  * H1 * W1;
    const size_t nPool3 = (size_t)B * 32 * H3 * W3;
    const size_t nI2p2a = (size_t)B * 8  * H3 * W3;
    const size_t nI2p3a = (size_t)B * 4  * H3 * W3;
    const size_t nI1p2a = (size_t)B * 8  * H2 * W2;
    const size_t nI1p3a = (size_t)B * 2  * H2 * W2;
    const size_t nPool2 = (size_t)B * 16 * H2 * W2;
    const size_t nI2out = (size_t)B * 44 * H3 * W3;

    float* pool1 = fb;                         // region R0 (largest early)
    float* c2o   = pool1 + nPool1;             // region R1
    float* c3o   = c2o + nC2;                  // region R2 (85M floats)
    float* pool2 = pool1;                      // reuse R0 (pool1 dead after c2)
    float* i1p2a = c2o;                        // reuse R1 (c2o dead after c3)
    float* i1p3a = i1p2a + nI1p2a;
    float* i1mp  = i1p3a + nI1p3a;
    float* i1out = c3o;                        // reuse R2 (c3o dead after pool2)
    float* pool3 = c2o;                        // reuse R1 (i1 temps dead)
    float* i2p2a = pool3 + nPool3;
    float* i2p3a = i2p2a + nI2p2a;
    float* i2mp  = i2p3a + nI2p3a;
    float* i2out = pool1 + nPool2;             // R0 after pool2 region
    float* gap   = i2out + nI2out;

    float* outp = (float*)d_out;               // [16,128] f32

    // ---- 1) sparsity bitmap ----
    zero_flags_kernel<<<(B * NV + 255) / 256, 256, 0, stream>>>(flags, B * NV);
    scatter_flags_kernel<<<(B * NT + 255) / 256, 256, 0, stream>>>(tags, flags);

    // ---- 2) fused scatter + conv1 + relu + pool1 ----
    {
        int total = B * 4 * H1 * W1;
        scatter_conv1_pool_kernel<<<(total + 255) / 256, 256, 0, stream>>>(
            flags, emb, c1_w, c1_b, pool1, H1, W1);
    }

    auto conv = [&](const float* in, const float* w, const float* bias, float* out,
                    int IC, int H, int W, int OC, int KH, int KW, int pad,
                    int OCtot, int oc0, int relu) {
        dim3 grid((H * W + 31) / 32, B);
        conv_wmma_kernel<<<grid, 32, 0, stream>>>(in, w, bias, out,
                                                  IC, H, W, OC, KH, KW, pad,
                                                  OCtot, oc0, relu);
    };
    auto pool = [&](const float* in, float* out, int C, int H, int W,
                    int Ho, int Wo, int s) {
        int total = B * C * Ho * Wo;
        maxpool3_kernel<<<(total + 255) / 256, 256, 0, stream>>>(
            in, out, B * C, H, W, Ho, Wo, s);
    };

    // ---- 3) c2 (1x1, 4->8), c3 (3x3, 8->16), pool2 ----
    conv(pool1, c2_w, c2_b, c2o, 4, H1, W1, 8, 1, 1, 0, 8, 0, 0);
    conv(c2o, c3_w, c3_b, c3o, 8, H1, W1, 16, 3, 3, 1, 16, 0, 0);
    pool(c3o, pool2, 16, H1, W1, H2, W2, 3);

    // ---- 4) inception 1 (16 -> 8+16+4+4 = 32), concat via oc0 ----
    conv(pool2, i1p1w,  i1p1b,  i1out, 16, H2, W2, 8,  1, 1, 0, 32, 0,  1);
    conv(pool2, i1p2aw, i1p2ab, i1p2a, 16, H2, W2, 8,  1, 1, 0, 8,  0,  1);
    conv(i1p2a, i1p2bw, i1p2bb, i1out, 8,  H2, W2, 16, 3, 3, 1, 32, 8,  1);
    conv(pool2, i1p3aw, i1p3ab, i1p3a, 16, H2, W2, 2,  1, 1, 0, 2,  0,  1);
    conv(i1p3a, i1p3bw, i1p3bb, i1out, 2,  H2, W2, 4,  5, 5, 2, 32, 24, 1);
    pool(pool2, i1mp, 16, H2, W2, H2, W2, 1);
    conv(i1mp,  i1p4w,  i1p4b,  i1out, 16, H2, W2, 4,  1, 1, 0, 32, 28, 1);

    // ---- 5) pool3 ----
    pool(i1out, pool3, 32, H2, W2, H3, W3, 2);

    // ---- 6) inception 2 (32 -> 16+12+8+8 = 44) ----
    conv(pool3, i2p1w,  i2p1b,  i2out, 32, H3, W3, 16, 1, 1, 0, 44, 0,  1);
    conv(pool3, i2p2aw, i2p2ab, i2p2a, 32, H3, W3, 8,  1, 1, 0, 8,  0,  1);
    conv(i2p2a, i2p2bw, i2p2bb, i2out, 8,  H3, W3, 12, 3, 3, 1, 44, 16, 1);
    conv(pool3, i2p3aw, i2p3ab, i2p3a, 32, H3, W3, 4,  1, 1, 0, 4,  0,  1);
    conv(i2p3a, i2p3bw, i2p3bb, i2out, 4,  H3, W3, 8,  5, 5, 2, 44, 28, 1);
    pool(pool3, i2mp, 32, H3, W3, H3, W3, 1);
    conv(i2mp,  i2p4w,  i2p4b,  i2out, 32, H3, W3, 8,  1, 1, 0, 44, 36, 1);

    // ---- 7) global avg pool -> [16,44] ----
    gap_kernel<<<B * 44, 256, 0, stream>>>(i2out, gap, H3 * W3);

    // ---- 8) FC [16,44] x [128,44]^T + b -> d_out [16,128] ----
    fc_wmma_kernel<<<8, 32, 0, stream>>>(gap, fc_w, fc_b, outp);
}
